// FasterMoEOutputOnlyMoE_51462298141175
// MI455X (gfx1250) — matrix operand order repaired and checked by reading, not compiled, not run on hardware
//
#include <hip/hip_runtime.h>

// ---------------------------------------------------------------------------
// Switch-gate top-1 MoE FFN for MI455X (gfx1250, wave32, WMMA).
// B=8 S=2048 D=1024 H=4096 E=16 -> N=16384 tokens, cap=1024.
// bf16 WMMA (16x16x32) grouped expert GEMMs with 128x128 block tiles staged
// in LDS via double-buffered GLOBAL_LOAD_ASYNC_TO_LDS_B128 (ASYNCcnt path).
// ---------------------------------------------------------------------------

typedef __attribute__((ext_vector_type(16))) __bf16 v16bf;
typedef __attribute__((ext_vector_type(8)))  __bf16 v8bf;
typedef __attribute__((ext_vector_type(4)))  __bf16 v4bf;
typedef __attribute__((ext_vector_type(8)))  float  v8f;
typedef __attribute__((ext_vector_type(4)))  float  v4f;

#define N_TOK  16384
#define DMODEL 1024
#define HID    4096
#define NEXP   16
#define CAP    1024   // N_TOK / NEXP
#define PITCH  72     // LDS row pitch in bf16 elements (144 B = 9 * 16 B)

// ------------------------------------------------------- async copy helpers
__device__ __forceinline__ void wait_async0() {
#if __has_builtin(__builtin_amdgcn_s_wait_asynccnt)
  __builtin_amdgcn_s_wait_asynccnt(0);
#else
  asm volatile("s_wait_asynccnt 0x0" ::: "memory");
#endif
}

// Copy 64 contiguous bytes global -> LDS via 4 async b128 transfers.
// The instruction's imm offset applies to BOTH the LDS and global address
// (CDNA5 ISA), so one base pair serves all 4 chunks.
__device__ __forceinline__ void stage64(const __bf16* g, __bf16* l) {
  unsigned lds_off = (unsigned)(unsigned long long)l;  // low 32 bits = LDS addr
  asm volatile(
      "global_load_async_to_lds_b128 %0, %1, off\n\t"
      "global_load_async_to_lds_b128 %0, %1, off offset:16\n\t"
      "global_load_async_to_lds_b128 %0, %1, off offset:32\n\t"
      "global_load_async_to_lds_b128 %0, %1, off offset:48"
      :: "v"(lds_off), "v"(g) : "memory");
}

// ---------------------------------------------------------------- zero output
__global__ __launch_bounds__(256) void zero_kernel(float* __restrict__ out, int n4) {
  int i = blockIdx.x * 256 + threadIdx.x;
  if (i < n4) {
    v4f z; z[0] = 0.f; z[1] = 0.f; z[2] = 0.f; z[3] = 0.f;
    *(v4f*)(out + 4 * (size_t)i) = z;
  }
}

// ------------------------------------------------------------------- gating
__global__ __launch_bounds__(256) void gate_kernel(
    const float* __restrict__ x, const float* __restrict__ Wg,
    const float* __restrict__ bg, float* __restrict__ gate,
    int* __restrict__ idx) {
  const int lane  = threadIdx.x & 31;
  const int token = blockIdx.x * 8 + (threadIdx.x >> 5);
  const float* xp = x + (size_t)token * DMODEL;

  float acc[NEXP];
#pragma unroll
  for (int e = 0; e < NEXP; ++e) acc[e] = 0.f;

  for (int d = lane; d < DMODEL; d += 32) {
    float xv = xp[d];
    const float* wr = Wg + (size_t)d * NEXP;
#pragma unroll
    for (int e = 0; e < NEXP; ++e) acc[e] = fmaf(xv, wr[e], acc[e]);
  }
#pragma unroll
  for (int off = 16; off >= 1; off >>= 1) {
#pragma unroll
    for (int e = 0; e < NEXP; ++e) acc[e] += __shfl_xor(acc[e], off);
  }
  if (lane == 0) {
    float m = acc[0] + bg[0];
    int best = 0;
#pragma unroll
    for (int e = 1; e < NEXP; ++e) {
      float z = acc[e] + bg[e];
      if (z > m) { m = z; best = e; }   // strict > keeps first max (argmax)
    }
    float s = 0.f;
#pragma unroll
    for (int e = 0; e < NEXP; ++e) s += __expf(acc[e] + bg[e] - m);
    gate[token] = 1.f / s;              // exp(z_max - m)/s with z_max == m
    idx[token]  = best;
  }
}

// -------------------------------------------------- deterministic dispatch
__global__ __launch_bounds__(256) void assign_kernel(
    const int* __restrict__ idx, int* __restrict__ srcTok) {
  const int e   = blockIdx.x;
  const int tid = threadIdx.x;
  const int lane = tid & 31, wv = tid >> 5;

  for (int i = tid; i < CAP; i += 256) srcTok[e * CAP + i] = -1;
  __syncthreads();

  __shared__ int waveTot[8];
  int base = 0;
  for (int t0 = 0; t0 < N_TOK; t0 += 256) {
    const int t = t0 + tid;
    const bool pred = (idx[t] == e);
    unsigned mask = (unsigned)__ballot(pred);   // wave32: low 32 bits
    if (lane == 0) waveTot[wv] = __popc(mask);
    __syncthreads();
    int wbase = 0, tot = 0;
#pragma unroll
    for (int w = 0; w < 8; ++w) {
      int c = waveTot[w];
      if (w < wv) wbase += c;
      tot += c;
    }
    if (pred) {
      int p = base + wbase + __popc(mask & ((1u << lane) - 1u));
      if (p < CAP) srcTok[e * CAP + p] = t;     // overflow tokens dropped
    }
    base += tot;
    __syncthreads();
  }
}

// ------------------------------------------------- dispatch x -> bf16 Abuf
__global__ __launch_bounds__(256) void build_a_kernel(
    const float* __restrict__ x, const int* __restrict__ srcTok,
    __bf16* __restrict__ Abuf) {
  const int row = blockIdx.x;
  const int i   = threadIdx.x * 4;
  const int tok = srcTok[row];
  v4bf o;
  if (tok >= 0) {
    v4f xv = *(const v4f*)(x + (size_t)tok * DMODEL + i);
#pragma unroll
    for (int j = 0; j < 4; ++j) o[j] = (__bf16)xv[j];
  } else {
#pragma unroll
    for (int j = 0; j < 4; ++j) o[j] = (__bf16)0.f;
  }
  *(v4bf*)(Abuf + (size_t)row * DMODEL + i) = o;
}

// ----------------------------------- fp32 -> bf16 transpose of expert weights
// src: [E][K, Nn] row-major fp32.  dst: [E][Nn, K] row-major bf16.
__global__ __launch_bounds__(256) void transpose_cast_kernel(
    const float* __restrict__ src, __bf16* __restrict__ dst, int K, int Nn) {
  __shared__ __bf16 tile[32][33];
  const int tx = threadIdx.x, ty = threadIdx.y;
  const int n0 = blockIdx.x * 32, k0 = blockIdx.y * 32;
  const size_t eoff = (size_t)blockIdx.z * (size_t)K * (size_t)Nn;
  const float* s = src + eoff;
  __bf16* d = dst + eoff;
#pragma unroll
  for (int j = 0; j < 4; ++j)
    tile[ty + 8 * j][tx] = (__bf16)s[(size_t)(k0 + ty + 8 * j) * Nn + (n0 + tx)];
  __syncthreads();
#pragma unroll
  for (int j = 0; j < 4; ++j)
    d[(size_t)(n0 + ty + 8 * j) * K + (k0 + tx)] = tile[tx][ty + 8 * j];
}

// --------------------------------------------------------------- GEMM core
// MODE 1: Hout = relu(A.B^T + b1)  (bf16 out, [N_TOK, HID])
// MODE 2: out[tok] = (A.B^T + b2) * gate[tok], scattered via srcTok.
// A: [rows, KTOT] bf16;  Bt: [E][NTOT, KTOT] bf16 (K contiguous per n).
// Block 256 thr = 8 waves (4M x 2N), block tile 128x128, wave tile 32x64.
// K pipelined in steps of 64 through double-buffered async LDS tiles.
template <int KTOT, int MODE>
__global__ __launch_bounds__(256) void moe_gemm_kernel(
    const __bf16* __restrict__ A, const __bf16* __restrict__ Bt,
    const float* __restrict__ bias, __bf16* __restrict__ Hout,
    const int* __restrict__ srcTok, const float* __restrict__ gate,
    float* __restrict__ out) {
  constexpr int NTOT = (MODE == 1) ? HID : DMODEL;
  __shared__ __bf16 sA[2][128][PITCH];
  __shared__ __bf16 sB[2][128][PITCH];

  const int tid  = threadIdx.x;
  const int lane = tid & 31;
  const int wv   = tid >> 5;
  const int la   = lane & 15;
  const int hb   = lane >> 4;
  const int mw   = (wv & 3) * 32;          // wave M offset in block tile
  const int nw   = (wv >> 2) * 64;         // wave N offset in block tile
  const int mBlk = blockIdx.x * 128;
  const int nBlk = blockIdx.y * 128;
  const int e    = mBlk >> 10;             // CAP = 1024

  // Staging coords: thread covers 64 contiguous bytes of one 128x64 tile row.
  const int sr = tid >> 1;                 // 0..127
  const int sc = (tid & 1) * 32;           // 0 or 32 (elements)
  const __bf16* gA = A + (size_t)(mBlk + sr) * KTOT + sc;
  const __bf16* gB = Bt + (size_t)e * DMODEL * HID + (size_t)(nBlk + sr) * KTOT + sc;

  v8f acc[2][4];
  float bv[4];
#pragma unroll
  for (int ti = 0; ti < 4; ++ti) {
    bv[ti] = bias[e * NTOT + nBlk + nw + ti * 16 + la];
#pragma unroll
    for (int mi = 0; mi < 2; ++mi)
#pragma unroll
      for (int i = 0; i < 8; ++i) acc[mi][ti][i] = 0.f;
  }

  // Prologue: stage first K-step.
  stage64(gA, &sA[0][sr][sc]);
  stage64(gB, &sB[0][sr][sc]);
  wait_async0();
  __syncthreads();

  int cur = 0;
  for (int kk = 0; kk < KTOT; kk += 64) {
    const int nxt = kk + 64;
    if (nxt < KTOT) {                      // issue next-step DMA before compute
      stage64(gA + nxt, &sA[cur ^ 1][sr][sc]);
      stage64(gB + nxt, &sB[cur ^ 1][sr][sc]);
    }
#pragma unroll
    for (int kj = 0; kj < 2; ++kj) {       // two K=32 sub-steps
      union { v16bf v; v8bf h[2]; } af[2], bfr[4];
#pragma unroll
      for (int mi = 0; mi < 2; ++mi) {
        const __bf16* p = &sA[cur][mw + mi * 16 + la][kj * 32 + hb * 8];
        af[mi].h[0] = *(const v8bf*)p;
        af[mi].h[1] = *(const v8bf*)(p + 16);
      }
#pragma unroll
      for (int ti = 0; ti < 4; ++ti) {
        const __bf16* p = &sB[cur][nw + ti * 16 + la][kj * 32 + hb * 8];
        bfr[ti].h[0] = *(const v8bf*)p;
        bfr[ti].h[1] = *(const v8bf*)(p + 16);
      }
#pragma unroll
      for (int mi = 0; mi < 2; ++mi)
#pragma unroll
        for (int ti = 0; ti < 4; ++ti)
          acc[mi][ti] = __builtin_amdgcn_wmma_f32_16x16x32_bf16(
              false, af[mi].v, false, bfr[ti].v, (short)0, acc[mi][ti], false, false);
    }
    if (nxt < KTOT) {
      wait_async0();
      __syncthreads();
      cur ^= 1;
    }
  }

  // Epilogues.
  if constexpr (MODE == 1) {
#pragma unroll
    for (int mi = 0; mi < 2; ++mi) {
#pragma unroll
      for (int ti = 0; ti < 4; ++ti) {
        const int col = nBlk + nw + ti * 16 + la;
#pragma unroll
        for (int v = 0; v < 8; ++v) {
          const int row = mBlk + mw + mi * 16 + hb * 8 + v;
          float r = acc[mi][ti][v] + bv[ti];
          r = r > 0.f ? r : 0.f;
          Hout[(size_t)row * HID + col] = (__bf16)r;
        }
      }
    }
  } else {
#pragma unroll
    for (int mi = 0; mi < 2; ++mi) {
#pragma unroll
      for (int v = 0; v < 8; ++v) {
        const int row = mBlk + mw + mi * 16 + hb * 8 + v;
        const int tok = srcTok[row];
        if (tok >= 0) {
          const float g = gate[tok];
#pragma unroll
          for (int ti = 0; ti < 4; ++ti) {
            const int col = nBlk + nw + ti * 16 + la;
            out[(size_t)tok * DMODEL + col] = (acc[mi][ti][v] + bv[ti]) * g;
          }
        }
      }
    }
  }
}

// ---------------------------------------------------------------------------
extern "C" void kernel_launch(void* const* d_in, const int* in_sizes, int n_in,
                              void* d_out, int out_size, void* d_ws, size_t ws_size,
                              hipStream_t stream) {
  const float* x  = (const float*)d_in[0];
  const float* Wg = (const float*)d_in[1];
  const float* bg = (const float*)d_in[2];
  const float* W1 = (const float*)d_in[3];
  const float* b1 = (const float*)d_in[4];
  const float* W2 = (const float*)d_in[5];
  const float* b2 = (const float*)d_in[6];
  float* out = (float*)d_out;

  // Workspace carve-out (~417 MB).
  char* p = (char*)d_ws;
  auto carve = [&](size_t bytes) {
    void* r = (void*)p;
    p += (bytes + 255) & ~(size_t)255;
    return r;
  };
  int*    idx    = (int*)carve((size_t)N_TOK * 4);
  float*  gate   = (float*)carve((size_t)N_TOK * 4);
  int*    srcTok = (int*)carve((size_t)NEXP * CAP * 4);
  __bf16* Abuf   = (__bf16*)carve((size_t)N_TOK * DMODEL * 2);
  __bf16* W1T    = (__bf16*)carve((size_t)NEXP * DMODEL * HID * 2);
  __bf16* W2T    = (__bf16*)carve((size_t)NEXP * DMODEL * HID * 2);
  __bf16* Hbuf   = (__bf16*)carve((size_t)N_TOK * HID * 2);

  {  // 1. Zero output (dropped tokens must be 0).
    int n4 = N_TOK * DMODEL / 4;
    zero_kernel<<<(n4 + 255) / 256, 256, 0, stream>>>(out, n4);
  }
  // 2. Gate: softmax top-1.
  gate_kernel<<<N_TOK / 8, 256, 0, stream>>>(x, Wg, bg, gate, idx);
  // 3. Deterministic per-expert slot assignment (cumsum order).
  assign_kernel<<<NEXP, 256, 0, stream>>>(idx, srcTok);
  // 4. Dispatch tokens -> bf16 expert buffers.
  build_a_kernel<<<N_TOK, 256, 0, stream>>>(x, srcTok, Abuf);
  // 5. fp32 -> bf16 transposed weights (K-contiguous B fragments).
  transpose_cast_kernel<<<dim3(HID / 32, DMODEL / 32, NEXP), dim3(32, 8), 0, stream>>>(
      W1, W1T, DMODEL, HID);
  transpose_cast_kernel<<<dim3(DMODEL / 32, HID / 32, NEXP), dim3(32, 8), 0, stream>>>(
      W2, W2T, HID, DMODEL);
  // 6. Expert GEMMs (bf16 WMMA, async-LDS double-buffered pipeline).
  moe_gemm_kernel<DMODEL, 1><<<dim3(N_TOK / 128, HID / 128), 256, 0, stream>>>(
      Abuf, W1T, b1, Hbuf, nullptr, nullptr, nullptr);
  moe_gemm_kernel<HID, 2><<<dim3(N_TOK / 128, DMODEL / 128), 256, 0, stream>>>(
      Hbuf, W2T, b2, nullptr, srcTok, gate, out);
}
